// SpectralAttention_75393855914099
// MI455X (gfx1250) — matrix operand order repaired and checked by reading, not compiled, not run on hardware
//
#include <hip/hip_runtime.h>

// ---------------------------------------------------------------------------
// Problem constants (match reference: B=4, T=2048, C=1024, H=16, d=64)
// ---------------------------------------------------------------------------
#define B_   4
#define T_   2048
#define C_   1024
#define H_   16
#define D_   64
#define NT_  (B_ * T_)        // 8192 tokens
#define NEGV (-1e9f)

#ifndef __has_builtin
#define __has_builtin(x) 0
#endif

#if __has_builtin(__builtin_amdgcn_global_load_async_to_lds_b128) && \
    __has_builtin(__builtin_amdgcn_s_wait_asynccnt)
#define USE_ASYNC_LDS 1
#else
#define USE_ASYNC_LDS 0
#endif

typedef __attribute__((ext_vector_type(16))) _Float16 v16h;
typedef __attribute__((ext_vector_type(8)))  float    v8f;
typedef __attribute__((ext_vector_type(4)))  int      i32x4;

union FragU {
  v16h     h;
  uint4    q[2];
  _Float16 e[16];
};

#if USE_ASYNC_LDS
// GLOBAL_LOAD_ASYNC_TO_LDS_B128: copies 16 B global -> LDS, tracked by
// ASYNCcnt (CDNA5 ISA). INST_OFFSET is added to BOTH
// the global and LDS addresses. Builtin expects (i32x4 addrspace(1)*,
// i32x4 addrspace(3)*, imm offset, imm cpol) per hipcc's diagnostic.
typedef __attribute__((address_space(1))) i32x4 gi32x4;
typedef __attribute__((address_space(3))) i32x4 li32x4;

__device__ __forceinline__ void async_cp_b128(const _Float16* g, _Float16* l) {
  gi32x4* gp = (gi32x4*)g;
  li32x4* lp = (li32x4*)l;
  __builtin_amdgcn_global_load_async_to_lds_b128(gp, lp, 0, 0);
  __builtin_amdgcn_global_load_async_to_lds_b128(gp, lp, 16, 0);
  __builtin_amdgcn_global_load_async_to_lds_b128(gp, lp, 32, 0);
  __builtin_amdgcn_global_load_async_to_lds_b128(gp, lp, 48, 0);
}
__device__ __forceinline__ void async_wait0() {
  __builtin_amdgcn_s_wait_asynccnt(0);
}
#else
__device__ __forceinline__ void async_cp_b128(const _Float16* g, _Float16* l) {
  const uint4* s = reinterpret_cast<const uint4*>(g);
  uint4* d = reinterpret_cast<uint4*>(l);
  d[0] = s[0]; d[1] = s[1]; d[2] = s[2]; d[3] = s[3];
}
__device__ __forceinline__ void async_wait0() {}
#endif

// ---------------------------------------------------------------------------
// WMMA fragment loaders from LDS (layouts per CDNA5 ISA 7.12.2, wave32)
//
// A 16x32 f16: lane l holds row M = l&15; two contiguous 16B chunks at
// k-offsets koff and koff+16 halves, koff = (l>>4)*8.
// B 32x16 f16: lane l holds col N = l&15; two contiguous 16B chunks at
// k-offsets koff and koff+16 halves, koff = (l>>4)*8 (same as A).
// ---------------------------------------------------------------------------
__device__ __forceinline__ v16h ld_frag_a(const _Float16* base, int stride,
                                          int m0, int kbase, int lane) {
  int row = m0 + (lane & 15);
  int ko  = kbase + ((lane >> 4) << 3);
  const _Float16* p = base + row * stride + ko;
  FragU f;
  f.q[0] = *reinterpret_cast<const uint4*>(p);
  f.q[1] = *reinterpret_cast<const uint4*>(p + 16);
  return f.h;
}

__device__ __forceinline__ v16h ld_frag_b(const _Float16* base, int stride,
                                          int n0, int kbase, int lane) {
  int col = n0 + (lane & 15);
  int ko  = kbase + ((lane >> 4) << 3);
  const _Float16* p = base + col * stride + ko;
  FragU f;
  f.q[0] = *reinterpret_cast<const uint4*>(p);
  f.q[1] = *reinterpret_cast<const uint4*>(p + 16);
  return f.h;
}

__device__ __forceinline__ v8f wmma16(v16h a, v16h b, v8f c) {
  // D(16x16 f32) = A(16x32 f16) * B(32x16 f16) + C
  return __builtin_amdgcn_wmma_f32_16x16x32_f16(
      /*neg_a=*/false, a, /*neg_b=*/false, b,
      /*c_mod=*/(short)0, c, /*reuse_a=*/false, /*reuse_b=*/false);
}

__device__ __forceinline__ void cvt16_f32_to_f16(const float* __restrict__ src,
                                                 _Float16* dst) {
  const float4* s = reinterpret_cast<const float4*>(src);
  float4 f0 = s[0], f1 = s[1], f2 = s[2], f3 = s[3];
  FragU u;
  u.e[0]=(_Float16)f0.x;  u.e[1]=(_Float16)f0.y;  u.e[2]=(_Float16)f0.z;  u.e[3]=(_Float16)f0.w;
  u.e[4]=(_Float16)f1.x;  u.e[5]=(_Float16)f1.y;  u.e[6]=(_Float16)f1.z;  u.e[7]=(_Float16)f1.w;
  u.e[8]=(_Float16)f2.x;  u.e[9]=(_Float16)f2.y;  u.e[10]=(_Float16)f2.z; u.e[11]=(_Float16)f2.w;
  u.e[12]=(_Float16)f3.x; u.e[13]=(_Float16)f3.y; u.e[14]=(_Float16)f3.z; u.e[15]=(_Float16)f3.w;
  uint4* d = reinterpret_cast<uint4*>(dst);
  d[0] = u.q[0];
  d[1] = u.q[1];
}

// ---------------------------------------------------------------------------
// Projection GEMM: out[m,n] = sum_k A[m,k] * W[n,k] + bias[n]
//   MODE 0: A = f32 x [8192,1024], out = f16 in [B,H,T,64] layout (QKV)
//   MODE 1: A = f16 y [8192,1024], out = f32 row-major [8192,1024] (final)
// Block: 256 threads (8 waves), tile 128(M) x 128(N), K-step 64 (2 WMMA
// k-steps per barrier pair -> 16 WMMAs / wave / stage).
// ---------------------------------------------------------------------------
template<int MODE>
__global__ __launch_bounds__(256)
void gemm_proj(const void* __restrict__ Ain, const float* __restrict__ W,
               const float* __restrict__ bias, void* __restrict__ outp) {
  __shared__ __align__(16) _Float16 As[128 * 72];   // 64 k + 8 pad
  __shared__ __align__(16) _Float16 Bs[128 * 72];

  const int tid  = threadIdx.x;
  const int lane = tid & 31;
  const int wid  = tid >> 5;
  const int wm   = (wid & 3) * 32;   // wave M offset in tile
  const int wn   = (wid >> 2) * 64;  // wave N offset in tile
  const int sr   = tid >> 1;         // staging row 0..127
  const int sc   = (tid & 1) << 5;   // staging k: 0 or 32

  const size_t mBase = (size_t)blockIdx.y * 128;
  const size_t nBase = (size_t)blockIdx.x * 128;

  const v8f vz = {0.f, 0.f, 0.f, 0.f, 0.f, 0.f, 0.f, 0.f};
  v8f acc[2][4];
#pragma unroll
  for (int i = 0; i < 2; ++i)
#pragma unroll
    for (int j = 0; j < 4; ++j) acc[i][j] = vz;

  for (int kk = 0; kk < C_; kk += 64) {
    __syncthreads();
    // ---- stage A tile (128 x 64 halves) ----
    if (MODE == 0) {
      const float* ap = (const float*)Ain + (mBase + sr) * C_ + kk + sc;
      cvt16_f32_to_f16(ap,      &As[sr * 72 + sc]);
      cvt16_f32_to_f16(ap + 16, &As[sr * 72 + sc + 16]);
      if (kk + 64 < C_) __builtin_prefetch(ap + 64, 0, 0);
    } else {
      const _Float16* ap = (const _Float16*)Ain + (mBase + sr) * C_ + kk + sc;
      async_cp_b128(ap, &As[sr * 72 + sc]);  // f16: async DMA to LDS
      if (kk + 64 < C_) __builtin_prefetch(ap + 64, 0, 0);
    }
    // ---- stage B tile: Bs[n][k] = W[n][k] (f32 -> f16) ----
    {
      const float* wp = W + (nBase + sr) * C_ + kk + sc;
      cvt16_f32_to_f16(wp,      &Bs[sr * 72 + sc]);
      cvt16_f32_to_f16(wp + 16, &Bs[sr * 72 + sc + 16]);
      if (kk + 64 < C_) __builtin_prefetch(wp + 64, 0, 0);
    }
    if (MODE == 1) async_wait0();
    __syncthreads();

    // ---- two 16x16x32 k-steps per stage ----
#pragma unroll
    for (int ks = 0; ks < 2; ++ks) {
      v16h a0 = ld_frag_a(As, 72, wm,      ks * 32, lane);
      v16h a1 = ld_frag_a(As, 72, wm + 16, ks * 32, lane);
#pragma unroll
      for (int ns = 0; ns < 4; ++ns) {
        v16h bf = ld_frag_b(Bs, 72, wn + ns * 16, ks * 32, lane);
        acc[0][ns] = wmma16(a0, bf, acc[0][ns]);
        acc[1][ns] = wmma16(a1, bf, acc[1][ns]);
      }
    }
  }

  // ---- epilogue: C layout -> VGPR r: (M = r + 8*(lane>=16), N = lane&15)
  const int mlo = (lane >> 4) << 3;
  const int nlo = lane & 15;
#pragma unroll
  for (int ms = 0; ms < 2; ++ms) {
#pragma unroll
    for (int ns = 0; ns < 4; ++ns) {
      const int mg0 = (int)mBase + wm + ms * 16 + mlo;
      const int ng  = (int)nBase + wn + ns * 16 + nlo;
      const float bv = bias[ng];
#pragma unroll
      for (int r = 0; r < 8; ++r) {
        const float v = acc[ms][ns][r] + bv;
        const int m = mg0 + r;
        if (MODE == 0) {
          const int bb = m >> 11, tt = m & (T_ - 1);
          const int hh = ng >> 6, dd = ng & 63;
          ((_Float16*)outp)[(((size_t)bb * H_ + hh) * T_ + tt) * D_ + dd] =
              (_Float16)v;
        } else {
          ((float*)outp)[(size_t)m * C_ + ng] = v;
        }
      }
    }
  }
}

// ---------------------------------------------------------------------------
// Fused causal attention (flash-style, online softmax).
// Grid: (T/64, B*H). Block: 128 threads = 4 waves; wave w owns 16 query rows.
// Q/K/V are f16 [B,H,T,64]; output Y is f16 in [B,T,H*64] = [B,T,C] layout
// (already transposed/merged for the final projection).
// Q and K tiles are staged with GLOBAL_LOAD_ASYNC_TO_LDS_B128 when available.
// ---------------------------------------------------------------------------
__global__ __launch_bounds__(128)
void attn_fused(const _Float16* __restrict__ Q, const _Float16* __restrict__ Kb,
                const _Float16* __restrict__ V, _Float16* __restrict__ Y) {
  __shared__ __align__(16) _Float16 Qs[64 * 72];      // [q][d]
  __shared__ __align__(16) _Float16 Ks[64 * 72];      // [key][d]  (B-frag, QK^T)
  __shared__ __align__(16) _Float16 Vt[64 * 72];      // [d][key]  (B-frag, PV)
  __shared__ __align__(16) _Float16 Ps[4 * 16 * 72];  // per-wave P [q][key]

  const int tid  = threadIdx.x;
  const int lane = tid & 31;
  const int wid  = tid >> 5;
  const int iq   = blockIdx.x;       // query block (64 rows)
  const int bh   = blockIdx.y;
  const int b    = bh >> 4;
  const int h    = bh & 15;

  const size_t head = ((size_t)b * H_ + h) * (size_t)T_ * D_;
  const _Float16* Qg = Q  + head + (size_t)iq * 64 * D_;
  const _Float16* Kg = Kb + head;
  const _Float16* Vg = V  + head;

  const int srow = tid >> 1;          // staging row 0..63
  const int scb  = (tid & 1) << 5;    // staging d: 0 or 32

  // ---- stage Q tile (64 x 64) once; completion covered by the per-
  //      iteration async_wait0() before the compute barrier ----
  async_cp_b128(Qg + srow * D_ + scb, &Qs[srow * 72 + scb]);

  const int mlo = (lane >> 4) << 3;  // row offset inside 16x16 C tile
  const int nlo = lane & 15;

  const v8f vz = {0.f, 0.f, 0.f, 0.f, 0.f, 0.f, 0.f, 0.f};
  float rowmax[8], rowsum[8];
  v8f o[4];
#pragma unroll
  for (int r = 0; r < 8; ++r) { rowmax[r] = -3.0e38f; rowsum[r] = 0.f; }
#pragma unroll
  for (int n = 0; n < 4; ++n) o[n] = vz;

  const float scale = 0.125f;  // d^-1/2 = 1/8

  for (int j = 0; j <= iq; ++j) {
    __syncthreads();
    // ---- stage K (async, row-major) and V (sync, transposed) tiles ----
    {
      async_cp_b128(Kg + ((size_t)j * 64 + srow) * D_ + scb,
                    &Ks[srow * 72 + scb]);

      const uint4* sv = reinterpret_cast<const uint4*>(
          Vg + ((size_t)j * 64 + srow) * D_ + scb);
      FragU u0, u1;
      u0.q[0] = sv[0]; u0.q[1] = sv[1];
      u1.q[0] = sv[2]; u1.q[1] = sv[3];
#pragma unroll
      for (int ii = 0; ii < 16; ++ii) Vt[(scb + ii) * 72 + srow]      = u0.e[ii];
#pragma unroll
      for (int ii = 0; ii < 16; ++ii) Vt[(scb + 16 + ii) * 72 + srow] = u1.e[ii];
    }
    async_wait0();     // Q (first iter) + K asyncs complete for this wave
    __syncthreads();   // ... and are visible to all waves

    // ---- S = (Q K^T) : 16 queries x 64 keys per wave ----
    v8f s[4];
#pragma unroll
    for (int n = 0; n < 4; ++n) s[n] = vz;
#pragma unroll
    for (int ks = 0; ks < 2; ++ks) {
      v16h a = ld_frag_a(Qs, 72, wid * 16, ks * 32, lane);
#pragma unroll
      for (int ns = 0; ns < 4; ++ns) {
        v16h bf = ld_frag_b(Ks, 72, ns * 16, ks * 32, lane);
        s[ns] = wmma16(a, bf, s[ns]);
      }
    }

    // ---- scale + causal mask (mask only on the diagonal block) ----
    if (j == iq) {
#pragma unroll
      for (int ns = 0; ns < 4; ++ns)
#pragma unroll
        for (int r = 0; r < 8; ++r) {
          const int qrow = wid * 16 + mlo + r;
          const int kcol = ns * 16 + nlo;
          const float v = s[ns][r] * scale;
          s[ns][r] = (kcol > qrow) ? NEGV : v;
        }
    } else {
#pragma unroll
      for (int ns = 0; ns < 4; ++ns)
#pragma unroll
        for (int r = 0; r < 8; ++r) s[ns][r] *= scale;
    }

    // ---- online softmax: rows live inside a 16-lane half of the wave ----
#pragma unroll
    for (int r = 0; r < 8; ++r) {
      float m = fmaxf(fmaxf(s[0][r], s[1][r]), fmaxf(s[2][r], s[3][r]));
      m = fmaxf(m, __shfl_xor(m, 8));
      m = fmaxf(m, __shfl_xor(m, 4));
      m = fmaxf(m, __shfl_xor(m, 2));
      m = fmaxf(m, __shfl_xor(m, 1));
      const float mnew  = fmaxf(rowmax[r], m);
      const float alpha = __expf(rowmax[r] - mnew);
      rowmax[r] = mnew;
      float p0 = __expf(s[0][r] - mnew);
      float p1 = __expf(s[1][r] - mnew);
      float p2 = __expf(s[2][r] - mnew);
      float p3 = __expf(s[3][r] - mnew);
      s[0][r] = p0; s[1][r] = p1; s[2][r] = p2; s[3][r] = p3;
      float rs = p0 + p1 + p2 + p3;
      rs += __shfl_xor(rs, 8);
      rs += __shfl_xor(rs, 4);
      rs += __shfl_xor(rs, 2);
      rs += __shfl_xor(rs, 1);
      rowsum[r] = rowsum[r] * alpha + rs;
#pragma unroll
      for (int ns = 0; ns < 4; ++ns) o[ns][r] *= alpha;
    }

    // ---- re-layout P: C-frag -> LDS -> A-frag (wave-private, DS in-order) --
    _Float16* Pw = &Ps[wid * 16 * 72];
#pragma unroll
    for (int ns = 0; ns < 4; ++ns)
#pragma unroll
      for (int r = 0; r < 8; ++r)
        Pw[(mlo + r) * 72 + ns * 16 + nlo] = (_Float16)s[ns][r];

    // ---- O += P * V ----
#pragma unroll
    for (int ks = 0; ks < 2; ++ks) {
      v16h a = ld_frag_a(Pw, 72, 0, ks * 32, lane);
#pragma unroll
      for (int ns = 0; ns < 4; ++ns) {
        v16h bf = ld_frag_b(Vt, 72, ns * 16, ks * 32, lane);
        o[ns] = wmma16(a, bf, o[ns]);
      }
    }
  }

  // ---- normalize and store into [B,T,C] (head-merged) f16 buffer ----
#pragma unroll
  for (int ns = 0; ns < 4; ++ns)
#pragma unroll
    for (int r = 0; r < 8; ++r) {
      const float v = o[ns][r] / rowsum[r];
      const int tq = iq * 64 + wid * 16 + mlo + r;
      const int dd = ns * 16 + nlo;
      Y[((size_t)b * T_ + tq) * C_ + h * D_ + dd] = (_Float16)v;
    }
}

// ---------------------------------------------------------------------------
// Launch: 3x QKV projection GEMMs -> fused attention -> output projection.
// Workspace: q,k,v,y f16 buffers (4 x 16 MB = 64 MB).
// ---------------------------------------------------------------------------
extern "C" void kernel_launch(void* const* d_in, const int* in_sizes, int n_in,
                              void* d_out, int out_size, void* d_ws,
                              size_t ws_size, hipStream_t stream) {
  const float* x  = (const float*)d_in[0];
  const float* wq = (const float*)d_in[1];
  const float* bq = (const float*)d_in[2];
  const float* wk = (const float*)d_in[3];
  const float* bk = (const float*)d_in[4];
  const float* wv = (const float*)d_in[5];
  const float* bv = (const float*)d_in[6];
  const float* wo = (const float*)d_in[7];
  const float* bo = (const float*)d_in[8];

  const size_t elems = (size_t)NT_ * C_;  // 8192*1024 per buffer
  _Float16* qb = (_Float16*)d_ws;
  _Float16* kb = qb + elems;
  _Float16* vb = kb + elems;
  _Float16* yb = vb + elems;

  const dim3 gGemm(C_ / 128, NT_ / 128);  // (8, 64)
  gemm_proj<0><<<gGemm, 256, 0, stream>>>((const void*)x, wq, bq, (void*)qb);
  gemm_proj<0><<<gGemm, 256, 0, stream>>>((const void*)x, wk, bk, (void*)kb);
  gemm_proj<0><<<gGemm, 256, 0, stream>>>((const void*)x, wv, bv, (void*)vb);

  const dim3 gAttn(T_ / 64, B_ * H_);  // (32, 64)
  attn_fused<<<gAttn, 128, 0, stream>>>(qb, kb, vb, yb);

  gemm_proj<1><<<gGemm, 256, 0, stream>>>((const void*)yb, wo, bo, d_out);
}
